// MultiHeadSelfAttention_57672820851035
// MI455X (gfx1250) — matrix operand order repaired and checked by reading, not compiled, not run on hardware
//
#include <hip/hip_runtime.h>

typedef __bf16 bf16_t;
typedef __attribute__((ext_vector_type(16))) __bf16 v16bf;
typedef __attribute__((ext_vector_type(8)))  float  v8f;

// Problem constants (match reference)
#define Lq   2048
#define Nb   4
#define Ed   1024
#define Hh   16
#define Dd   64
#define LN   (Lq * Nb)      // 8192 rows
#define F3   (3 * Ed)       // 3072

// ---------------------------------------------------------------------------
// fp32 -> bf16 conversion (round via cast, RNE on clang/amdgcn)
// ---------------------------------------------------------------------------
__global__ __launch_bounds__(256) void cvt_f32_bf16(const float* __restrict__ in,
                                                    bf16_t* __restrict__ out, int n) {
    int i = blockIdx.x * blockDim.x + threadIdx.x;
    int stride = gridDim.x * blockDim.x;
    for (; i < n; i += stride) out[i] = (bf16_t)in[i];
}

// ---------------------------------------------------------------------------
// QKV projection: QKV[m, f] = sum_e X[m, e] * W[f, e]
// X: [LN, E] bf16, W: [3E, E] bf16.
// Q region (cols 0..1023): scaled by d^-0.5 and stored to QKV.
// K region (cols 1024..2047): stored to QKV.
// V region (cols 2048..3071): stored TRANSPOSED to Vt[n][h][c][l] so the
//   attention P*V B-tile becomes a contiguous per-lane load.
// One wave computes a 16x64 output tile (4 WMMA C tiles reusing one A tile).
// ---------------------------------------------------------------------------
__global__ __launch_bounds__(256) void qkv_gemm(const bf16_t* __restrict__ X,
                                                const bf16_t* __restrict__ W,
                                                bf16_t* __restrict__ QKV,
                                                bf16_t* __restrict__ Vt) {
    const int wave = threadIdx.x >> 5;
    const int lane = threadIdx.x & 31;
    const int half = lane >> 4;
    const int l16  = lane & 15;

    const int row0 = blockIdx.y * 128 + wave * 16;
    const int col0 = blockIdx.x * 64;

    v8f acc[4] = {};

    const bf16_t* arow = X + (size_t)(row0 + l16) * Ed;

    for (int k0 = 0; k0 < Ed; k0 += 32) {
        // A tile 16x32 bf16: a[0..15]=K(k0+half*16 ..)
        const bf16_t* pa = arow + k0 + half * 16;
        __builtin_prefetch(pa + 128, 0, 0);
        v16bf a;
#pragma unroll
        for (int i = 0; i < 16; ++i) a[i] = pa[i];

#pragma unroll
        for (int t = 0; t < 4; ++t) {
            // B tile 32x16: lane column = output col, contiguous in K (W row-major [F,E])
            const bf16_t* pb = W + (size_t)(col0 + t * 16 + l16) * Ed + k0 + half * 16;
            v16bf b;
#pragma unroll
            for (int i = 0; i < 16; ++i) b[i] = pb[i];
            acc[t] = __builtin_amdgcn_wmma_f32_16x16x32_bf16(
                false, a, false, b, (short)0, acc[t], false, false);
        }
    }

    if (col0 >= 2 * Ed) {
        // V region -> transposed store Vt[((n*H + h)*D + c) * L + l]
#pragma unroll
        for (int t = 0; t < 4; ++t)
#pragma unroll
            for (int r = 0; r < 8; ++r) {
                const int m = row0 + r + 8 * half;       // m = l*Nb + n
                const int l = m >> 2, n = m & 3;
                const int f = col0 - 2 * Ed + t * 16 + l16;
                const int h = f >> 6, c = f & 63;
                Vt[(((size_t)n * Hh + h) * Dd + c) * Lq + l] = (bf16_t)acc[t][r];
            }
    } else {
        // Q/K regions -> row-major store; fold softmax scale into Q (ref scales q)
        const float s = (col0 < Ed) ? 0.125f : 1.0f;
#pragma unroll
        for (int t = 0; t < 4; ++t)
#pragma unroll
            for (int r = 0; r < 8; ++r) {
                const int row = row0 + r + 8 * half;
                const int col = col0 + t * 16 + l16;
                QKV[(size_t)row * F3 + col] = (bf16_t)(acc[t][r] * s);
            }
    }
}

// ---------------------------------------------------------------------------
// Flash-style causal attention. One wave handles one (head, 16-query block).
// Q/K from QKV [(l*Nb + n), 3E] (Q pre-scaled); V from Vt[n][h][c][l].
// Output O: [L, Nb, E] bf16.
// ---------------------------------------------------------------------------
__global__ __launch_bounds__(256) void attn_fwd(const bf16_t* __restrict__ QKV,
                                                const bf16_t* __restrict__ Vt,
                                                bf16_t* __restrict__ O) {
    __shared__ bf16_t sh[8 * 16 * 32];   // per-wave 16x32 P staging (8 KB)

    const int wave = threadIdx.x >> 5;
    const int lane = threadIdx.x & 31;
    const int half = lane >> 4;
    const int l16  = lane & 15;

    const int wid  = blockIdx.x * 8 + wave;
    const int head = wid >> 7;          // 64 heads
    const int qb   = wid & 127;         // 128 query blocks of 16
    const int n    = head >> 4;         // batch
    const int h    = head & 15;         // head within batch
    const int q0   = qb * 16;

    // Load Q as two 16x32 A tiles (K dim = 64 channels); Q is pre-scaled.
    v16bf aq[2];
#pragma unroll
    for (int kk = 0; kk < 2; ++kk) {
        const bf16_t* pq = QKV + ((size_t)(q0 + l16) * Nb + n) * F3 + h * Dd + kk * 32 + half * 16;
        v16bf a;
#pragma unroll
        for (int i = 0; i < 16; ++i) a[i] = pq[i];
        aq[kk] = a;
    }

    // Per-head V panel base: Vt[((n*H + h)*D + c)*L + l]
    const bf16_t* vbase = Vt + ((size_t)n * Hh + h) * Dd * Lq;

    float mrow[8], lrow[8];
#pragma unroll
    for (int r = 0; r < 8; ++r) { mrow[r] = -3.0e38f; lrow[r] = 0.0f; }
    v8f accO[4] = {};

    bf16_t* myp = sh + wave * 512;
    const int jn = (q0 + 15) / 32;

    for (int j = 0; j <= jn; ++j) {
        const int m0 = j * 32;

        // S = Q * K^T for 32 keys -> two 16x16 score tiles
        v8f s0 = {}, s1 = {};
#pragma unroll
        for (int kk = 0; kk < 2; ++kk) {
            const bf16_t* pk0 = QKV + ((size_t)(m0 + l16) * Nb + n) * F3 + Ed + h * Dd + kk * 32 + half * 16;
            const bf16_t* pk1 = QKV + ((size_t)(m0 + 16 + l16) * Nb + n) * F3 + Ed + h * Dd + kk * 32 + half * 16;
            __builtin_prefetch(pk0 + (size_t)F3 * 32, 0, 0);
            v16bf b0, b1;
#pragma unroll
            for (int i = 0; i < 16; ++i) { b0[i] = pk0[i]; b1[i] = pk1[i]; }
            s0 = __builtin_amdgcn_wmma_f32_16x16x32_bf16(false, aq[kk], false, b0, (short)0, s0, false, false);
            s1 = __builtin_amdgcn_wmma_f32_16x16x32_bf16(false, aq[kk], false, b1, (short)0, s1, false, false);
        }

        const int key0 = m0 + l16;
        const int key1 = m0 + 16 + l16;

#pragma unroll
        for (int r = 0; r < 8; ++r) {
            const int row = q0 + r + 8 * half;
            float v0 = s0[r]; if (key0 > row) v0 = -10000.0f;
            float v1 = s1[r]; if (key1 > row) v1 = -10000.0f;

            // row max across 16 lanes (C tile stripes one row across lanes 0..15 / 16..31)
            float bm = fmaxf(v0, v1);
#pragma unroll
            for (int off = 1; off < 16; off <<= 1) bm = fmaxf(bm, __shfl_xor(bm, off, 16));

            const float nm    = fmaxf(mrow[r], bm);
            const float alpha = __expf(mrow[r] - nm);
            const float e0 = __expf(v0 - nm);
            const float e1 = __expf(v1 - nm);
            float rs = e0 + e1;
#pragma unroll
            for (int off = 1; off < 16; off <<= 1) rs += __shfl_xor(rs, off, 16);

            lrow[r] = lrow[r] * alpha + rs;
            mrow[r] = nm;
#pragma unroll
            for (int t = 0; t < 4; ++t) accO[t][r] *= alpha;

            // stash P (C layout) to LDS for C->A relayout
            myp[(r + 8 * half) * 32 + l16]      = (bf16_t)e0;
            myp[(r + 8 * half) * 32 + 16 + l16] = (bf16_t)e1;
        }

        asm volatile("s_wait_dscnt 0" ::: "memory");

        // Reload P as a 16x32 A tile
        const bf16_t* pr = myp + l16 * 32 + half * 16;
        v16bf pa;
#pragma unroll
        for (int i = 0; i < 16; ++i) pa[i] = pr[i];

        // accO += P * V : B tile b[i] = V[key m0+half*16+i][chan t*16+l16]
        // = Vt[(t*16+l16)*L + m0+half*16+i]  -> contiguous 32B per lane
#pragma unroll
        for (int t = 0; t < 4; ++t) {
            const bf16_t* pv = vbase + (size_t)(t * 16 + l16) * Lq + m0 + half * 16;
            v16bf bv;
#pragma unroll
            for (int i = 0; i < 16; ++i) bv[i] = pv[i];
            accO[t] = __builtin_amdgcn_wmma_f32_16x16x32_bf16(
                false, pa, false, bv, (short)0, accO[t], false, false);
        }
    }

    // epilogue: normalize and store O[l, n, h*64 + c]
#pragma unroll
    for (int t = 0; t < 4; ++t)
#pragma unroll
        for (int r = 0; r < 8; ++r) {
            const int lrowi = q0 + r + 8 * half;
            const float o = accO[t][r] / lrow[r];
            O[((size_t)lrowi * Nb + n) * Ed + h * Dd + t * 16 + l16] = (bf16_t)o;
        }
}

// ---------------------------------------------------------------------------
// Column reductions + final dot (out-projection collapsed by linearity:
// sum(out) = sum_e colsum_O[e] * (sum_f wout[f,e]) )
// ---------------------------------------------------------------------------
__global__ __launch_bounds__(256) void colsum_bf16(const bf16_t* __restrict__ A,
                                                   float* __restrict__ out,
                                                   int rows, int cols) {
    __shared__ float red[256];
    const int e = blockIdx.x;
    float s = 0.0f;
    for (int r = threadIdx.x; r < rows; r += 256) s += (float)A[(size_t)r * cols + e];
    red[threadIdx.x] = s; __syncthreads();
    for (int o = 128; o > 0; o >>= 1) {
        if (threadIdx.x < o) red[threadIdx.x] += red[threadIdx.x + o];
        __syncthreads();
    }
    if (threadIdx.x == 0) out[e] = red[0];
}

__global__ __launch_bounds__(256) void colsum_f32(const float* __restrict__ A,
                                                  float* __restrict__ out,
                                                  int rows, int cols) {
    __shared__ float red[256];
    const int e = blockIdx.x;
    float s = 0.0f;
    for (int r = threadIdx.x; r < rows; r += 256) s += A[(size_t)r * cols + e];
    red[threadIdx.x] = s; __syncthreads();
    for (int o = 128; o > 0; o >>= 1) {
        if (threadIdx.x < o) red[threadIdx.x] += red[threadIdx.x + o];
        __syncthreads();
    }
    if (threadIdx.x == 0) out[e] = red[0];
}

__global__ __launch_bounds__(256) void final_dot(const float* __restrict__ a,
                                                 const float* __restrict__ b,
                                                 float* __restrict__ out, int n) {
    __shared__ float red[256];
    float s = 0.0f;
    for (int i = threadIdx.x; i < n; i += 256) s += a[i] * b[i];
    red[threadIdx.x] = s; __syncthreads();
    for (int o = 128; o > 0; o >>= 1) {
        if (threadIdx.x < o) red[threadIdx.x] += red[threadIdx.x + o];
        __syncthreads();
    }
    if (threadIdx.x == 0) out[0] = red[0];
}

// ---------------------------------------------------------------------------
extern "C" void kernel_launch(void* const* d_in, const int* in_sizes, int n_in,
                              void* d_out, int out_size, void* d_ws, size_t ws_size,
                              hipStream_t stream) {
    const float* x    = (const float*)d_in[0];   // [L, N, E]
    const float* wqkv = (const float*)d_in[1];   // [3E, E]
    const float* wout = (const float*)d_in[2];   // [E, E]
    float* out = (float*)d_out;

    // workspace carve-up (bytes)
    char* w = (char*)d_ws;
    bf16_t* Xbf  = (bf16_t*)w;                       w += (size_t)LN * Ed * 2;        // 16 MB
    bf16_t* Wbf  = (bf16_t*)w;                       w += (size_t)F3 * Ed * 2;        //  6 MB
    bf16_t* QKV  = (bf16_t*)w;                       w += (size_t)LN * F3 * 2;        // 48 MB
    bf16_t* Vt   = (bf16_t*)w;                       w += (size_t)Nb * Ed * Lq * 2;   // 16 MB
    bf16_t* Obf  = (bf16_t*)w;                       w += (size_t)LN * Ed * 2;        // 16 MB
    float*  csO  = (float*)w;                        w += (size_t)Ed * 4;
    float*  csW  = (float*)w;                        w += (size_t)Ed * 4;

    // 1) fp32 -> bf16
    cvt_f32_bf16<<<2048, 256, 0, stream>>>(x,    Xbf, in_sizes[0]);
    cvt_f32_bf16<<<2048, 256, 0, stream>>>(wqkv, Wbf, in_sizes[1]);

    // 2) QKV projection: [8192,1024] x [1024,3072]; Q scaled, V transposed
    qkv_gemm<<<dim3(F3 / 64, LN / 128), 256, 0, stream>>>(Xbf, Wbf, QKV, Vt);

    // 3) causal flash attention: 64 heads * 128 qblocks = 8192 waves / 8 per block
    attn_fwd<<<1024, 256, 0, stream>>>(QKV, Vt, Obf);

    // 4) collapsed output projection + global sum
    colsum_bf16<<<Ed, 256, 0, stream>>>(Obf, csO, LN, Ed);
    colsum_f32 <<<Ed, 256, 0, stream>>>(wout, csW, Ed, Ed);
    final_dot  <<<1, 256, 0, stream>>>(csO, csW, out, Ed);
}
